// BertSelfAttention_49409303773384
// MI455X (gfx1250) — matrix operand order repaired and checked by reading, not compiled, not run on hardware
//
#include <hip/hip_runtime.h>
#include <hip/hip_bf16.h>
#include <stdint.h>

// ---------------------------------------------------------------------------
// BERT self-attention block on gfx1250 (MI455X), bf16 WMMA everywhere,
// async-to-LDS staging for GEMM tiles, TDM tensor DMA for attention K tiles.
// B=8, S=1024, H=12, DH=64, D=768.
// ---------------------------------------------------------------------------

typedef __bf16 bf16_t;
typedef __attribute__((ext_vector_type(16))) __bf16 v16bf;
typedef __attribute__((ext_vector_type(8)))  __bf16 v8bf;   // 16-byte chunk
typedef __attribute__((ext_vector_type(8)))  float  v8f;
typedef unsigned int u32x4 __attribute__((ext_vector_type(4)));
typedef int          i32x8 __attribute__((ext_vector_type(8)));
typedef int          i32x4 __attribute__((ext_vector_type(4)));

#define NB 8
#define NS 1024
#define NH 12
#define NDH 64
#define ND 768
#define NM (NB * NS)   // 8192 rows

// TDM available? (this toolchain: 6-arg builtin; device pass only)
#if defined(__gfx1250__) && __has_builtin(__builtin_amdgcn_tensor_load_to_lds) && \
    __has_builtin(__builtin_amdgcn_s_wait_tensorcnt)
#define USE_TDM 1
#endif

__device__ __forceinline__ v8f wmma_bf16(v16bf a, v16bf b, v8f c) {
  // D = A(16x32) * B(32x16) + C(16x16 f32)
  return __builtin_amdgcn_wmma_f32_16x16x32_bf16(
      /*neg_a=*/false, a, /*neg_b=*/false, b,
      /*c_mod=*/(short)0, c, /*reuse_a=*/false, /*reuse_b=*/false);
}

// Async copy of two 16B chunks (32B) global -> LDS, ASYNCcnt-tracked.
// Per CDNA5 ISA: INST_OFFSET is added to BOTH the LDS dest and
// the global address, so one address pair covers offset 0 and 16.
__device__ __forceinline__ void async_cp32(uint32_t lds_off, uint64_t gaddr) {
#if defined(__HIP_DEVICE_COMPILE__)
  asm volatile(
      "global_load_async_to_lds_b128 %0, %1, off\n\t"
      "global_load_async_to_lds_b128 %0, %1, off offset:16"
      :: "v"(lds_off), "v"(gaddr) : "memory");
#else
  (void)lds_off; (void)gaddr;
#endif
}

__device__ __forceinline__ void async_wait0() {
#if defined(__HIP_DEVICE_COMPILE__)
  asm volatile("s_wait_asynccnt 0" ::: "memory");
#endif
}

__device__ __forceinline__ void ds_wait0() {
#if defined(__HIP_DEVICE_COMPILE__)
  asm volatile("s_wait_dscnt 0" ::: "memory");
#endif
}

// ---------------------------------------------------------------------------
// Kernel 0: fp32 -> bf16 conversion
// ---------------------------------------------------------------------------
__global__ __launch_bounds__(256)
void cvt_kernel(const float* __restrict__ src, bf16_t* __restrict__ dst, int n) {
  int i = blockIdx.x * 256 + threadIdx.x;
  if (i < n) dst[i] = (bf16_t)src[i];
}

// ---------------------------------------------------------------------------
// Kernel 1: QKV projection GEMM.  Y = X @ W^T + bias, written bf16 into
// [B,H,S,DH] layout, optionally scaled (Q gets 1/sqrt(DH) folded in).
// Workgroup: 128 threads (4 waves), 64x64 output tile, async-to-LDS staging.
// ---------------------------------------------------------------------------
__global__ __launch_bounds__(128)
void qkv_gemm_kernel(const bf16_t* __restrict__ X,
                     const bf16_t* __restrict__ W,
                     const float*  __restrict__ bias,
                     bf16_t* __restrict__ Out,
                     float scale) {
  __shared__ bf16_t As[64][40];   // [m_local][k_local], padded
  __shared__ bf16_t Bs[64][40];   // [n_local][k_local], padded

  const int m0   = blockIdx.x * 64;
  const int n0   = blockIdx.y * 64;
  const int tid  = threadIdx.x;
  const int wave = tid >> 5;
  const int lane = tid & 31;
  const int hrow = lane & 15;     // row for A-frag / col for B,C-frag
  const int hi   = lane >> 4;     // which 16-lane half

  v8f acc[4] = {};

  const int r = tid >> 1;
  const int c = (tid & 1) * 16;
  const uint32_t la = (uint32_t)(uintptr_t)&As[r][c];
  const uint32_t lb = (uint32_t)(uintptr_t)&Bs[r][c];

  for (int k0 = 0; k0 < ND; k0 += 32) {
    __syncthreads();
    // Async stage A and B tiles (32B per thread per tile), bypassing VGPRs.
    async_cp32(la, (uint64_t)(uintptr_t)(X + (size_t)(m0 + r) * ND + k0 + c));
    async_cp32(lb, (uint64_t)(uintptr_t)(W + (size_t)(n0 + r) * ND + k0 + c));
    async_wait0();
    __syncthreads();

    // A fragment (16x32 bf16): lanes 0-15 -> K {0..7,16..23}, 16-31 -> K {8..15,24..31}
    v16bf a;
    {
      const bf16_t* pa  = &As[wave * 16 + hrow][hi * 8];
      const bf16_t* pa2 = &As[wave * 16 + hrow][hi * 8 + 16];
#pragma unroll
      for (int j = 0; j < 8; j++) { a[j] = pa[j]; a[8 + j] = pa2[j]; }
    }
#pragma unroll
    for (int t = 0; t < 4; t++) {
      // B fragment (32x16): lanes 0-15 hold K {0..7,16..23}, lanes 16-31 hold K {8..15,24..31}
      v16bf b;
      const bf16_t* pb  = &Bs[t * 16 + hrow][hi * 8];
      const bf16_t* pb2 = &Bs[t * 16 + hrow][hi * 8 + 16];
#pragma unroll
      for (int j = 0; j < 8; j++) { b[j] = pb[j]; b[8 + j] = pb2[j]; }
      acc[t] = wmma_bf16(a, b, acc[t]);
    }
  }

  // Epilogue: C layout -> [B,H,S,DH] bf16 with bias and scale.
#pragma unroll
  for (int t = 0; t < 4; t++) {
    int n  = n0 + t * 16 + hrow;      // global N = h*64 + d
    float bv = bias[n];
    int h = n >> 6, d = n & 63;
#pragma unroll
    for (int rr = 0; rr < 8; rr++) {
      int m = m0 + wave * 16 + hi * 8 + rr;   // global M = b*1024 + s
      int b = m >> 10, s = m & 1023;
      float y = (acc[t][rr] + bv) * scale;
      Out[(((size_t)(b * NH + h) * NS + s) * NDH) + d] = (bf16_t)y;
    }
  }
}

// ---------------------------------------------------------------------------
// Kernel 2: flash attention.  One workgroup per (q-block of 64, h, b).
// K blocks staged via TDM tensor DMA (with LDS row padding via pad_enable);
// V staged transposed so PV B-fragments are contiguous LDS reads.
// ---------------------------------------------------------------------------
__global__ __launch_bounds__(128)
void attn_kernel(const bf16_t* __restrict__ Q,
                 const bf16_t* __restrict__ K,
                 const bf16_t* __restrict__ V,
                 const float*  __restrict__ mask,   // [B,1,1,S]
                 bf16_t* __restrict__ Ctx) {        // [B,S,768] bf16
  __shared__ bf16_t Ks[128][72];       // [key_local][d], row pad = 8 elems (16B)
  __shared__ bf16_t Vs[64][136];       // [d][key_local]  (transposed)
  __shared__ bf16_t Ps[4][16][40];     // per-wave P-tile scratch (16x32)

  const int b = blockIdx.z, h = blockIdx.y;
  const int q0 = blockIdx.x * 64;
  const int tid  = threadIdx.x;
  const int wave = tid >> 5;
  const int lane = tid & 31;
  const int hrow = lane & 15;
  const int hi   = lane >> 4;

  const bf16_t* Qbh = Q + (size_t)(b * NH + h) * NS * NDH;
  const bf16_t* Kbh = K + (size_t)(b * NH + h) * NS * NDH;
  const bf16_t* Vbh = V + (size_t)(b * NH + h) * NS * NDH;
  const float* maskb = mask + (size_t)b * NS;

  // Preload this wave's Q A-fragments (16 rows x 64, two K-chunks of 32).
  v16bf qf[2];
  {
    const bf16_t* qp = Qbh + (size_t)(q0 + wave * 16 + hrow) * NDH;
#pragma unroll
    for (int cc = 0; cc < 2; cc++) {
#pragma unroll
      for (int j = 0; j < 8; j++) {
        qf[cc][j]     = qp[cc * 32 + hi * 8 + j];
        qf[cc][8 + j] = qp[cc * 32 + hi * 8 + 16 + j];
      }
    }
  }

  float mrow[8], lrow[8];
  v8f o[4] = {};
#pragma unroll
  for (int rr = 0; rr < 8; rr++) { mrow[rr] = -3.0e38f; lrow[rr] = 0.0f; }

  for (int kb0 = 0; kb0 < NS; kb0 += 128) {
    __syncthreads();
#if defined(USE_TDM)
    if (wave == 0) {
      // TDM D#: 2D tensor [1024 x 64] (2B elems), tile [128 x 64] at row kb0,
      // destination Ks with 16B pad after every 128B row (stride 72 elems).
      uint64_t ga  = (uint64_t)(uintptr_t)(Kbh + (size_t)kb0 * NDH);
      uint32_t lds = (uint32_t)(uintptr_t)&Ks[0][0];
      u32x4 g0 = { 1u,                                  // count=1, is_restore=0
                   lds,                                 // lds_addr
                   (uint32_t)ga,                        // global_addr[31:0]
                   (uint32_t)(ga >> 32) | (2u << 30) }; // addr[56:32] | type=2
      i32x8 g1 = { (int)((1u << 16) |                   // data_size = 2 bytes
                         (1u << 20) |                   // pad_enable
                         (4u << 22) |                   // pad_interval: 32 DW (128B)
                         (3u << 25)),                   // pad_amount: 4 DW (16B)
                   (int)(64u << 16),                    // tensor_dim0 = 64
                   (int)(1024u << 16),                  // tensor_dim1 = 1024
                   (int)(64u << 16),                    // tile_dim0 = 64
                   128,                                 // tile_dim1 = 128
                   64,                                  // tensor_dim0_stride = 64
                   0, 0 };
      i32x4 gz4 = { 0, 0, 0, 0 };
      i32x8 gz8 = { 0, 0, 0, 0, 0, 0, 0, 0 };
      __builtin_amdgcn_tensor_load_to_lds(g0, g1, gz4, gz4, gz8, 0);
      __builtin_amdgcn_s_wait_tensorcnt(0);
    }
#else
    {
      int key = tid;
      const bf16_t* kp = Kbh + (size_t)(kb0 + key) * NDH;
#pragma unroll
      for (int cc = 0; cc < 8; cc++)
        *reinterpret_cast<v8bf*>(&Ks[key][cc * 8]) =
            *reinterpret_cast<const v8bf*>(kp + cc * 8);
    }
#endif
    // V transposed staging (all threads; independent LDS region).
    {
      int key = tid;
      const bf16_t* vp = Vbh + (size_t)(kb0 + key) * NDH;
#pragma unroll
      for (int d = 0; d < 64; d++) Vs[d][key] = vp[d];
    }
    __syncthreads();

    for (int sub = 0; sub < 4; sub++) {      // 32 keys per inner step
      // ---- scores: 16x32 tile = two 16x16 C-fragments -------------------
      v8f sc[2] = {};
#pragma unroll
      for (int f = 0; f < 2; f++) {
#pragma unroll
        for (int cc = 0; cc < 2; cc++) {
          v16bf kf;
          const bf16_t* pk = &Ks[sub * 32 + f * 16 + hrow][cc * 32 + hi * 8];
#pragma unroll
          for (int j = 0; j < 8; j++) { kf[j] = pk[j]; kf[8 + j] = pk[16 + j]; }
          sc[f] = wmma_bf16(qf[cc], kf, sc[f]);
        }
      }
      // ---- mask + running softmax stats ---------------------------------
      float bmax[8];
#pragma unroll
      for (int rr = 0; rr < 8; rr++) bmax[rr] = -3.0e38f;
#pragma unroll
      for (int f = 0; f < 2; f++) {
        float mv = maskb[kb0 + sub * 32 + f * 16 + hrow];
#pragma unroll
        for (int rr = 0; rr < 8; rr++) {
          sc[f][rr] += mv;
          bmax[rr] = fmaxf(bmax[rr], sc[f][rr]);
        }
      }
      // Reduce across the 16 lanes of each half (rows differ across halves!)
#pragma unroll
      for (int off = 1; off < 16; off <<= 1)
#pragma unroll
        for (int rr = 0; rr < 8; rr++)
          bmax[rr] = fmaxf(bmax[rr], __shfl_xor(bmax[rr], off, 32));

      float corr[8], psum[8];
#pragma unroll
      for (int rr = 0; rr < 8; rr++) {
        float mnew = fmaxf(mrow[rr], bmax[rr]);
        corr[rr] = __expf(mrow[rr] - mnew);
        mrow[rr] = mnew;
        psum[rr] = 0.0f;
      }
      // p = exp(s - m); stage P tile to LDS in C layout.
#pragma unroll
      for (int f = 0; f < 2; f++)
#pragma unroll
        for (int rr = 0; rr < 8; rr++) {
          float p = __expf(sc[f][rr] - mrow[rr]);
          psum[rr] += p;
          Ps[wave][hi * 8 + rr][f * 16 + hrow] = (bf16_t)p;
        }
#pragma unroll
      for (int off = 1; off < 16; off <<= 1)
#pragma unroll
        for (int rr = 0; rr < 8; rr++) psum[rr] += __shfl_xor(psum[rr], off, 32);
#pragma unroll
      for (int rr = 0; rr < 8; rr++) lrow[rr] = lrow[rr] * corr[rr] + psum[rr];
#pragma unroll
      for (int t = 0; t < 4; t++)
#pragma unroll
        for (int rr = 0; rr < 8; rr++) o[t][rr] *= corr[rr];

      // P C-layout -> A-layout via per-wave LDS round trip.
      ds_wait0();
      v16bf pa;
      {
        const bf16_t* pp  = &Ps[wave][hrow][hi * 8];
        const bf16_t* pp2 = &Ps[wave][hrow][hi * 8 + 16];
#pragma unroll
        for (int j = 0; j < 8; j++) { pa[j] = pp[j]; pa[8 + j] = pp2[j]; }
      }
      // ---- O += P x V ----------------------------------------------------
#pragma unroll
      for (int t = 0; t < 4; t++) {
        v16bf vf;
        const bf16_t* pv = &Vs[t * 16 + hrow][sub * 32 + hi * 8];
#pragma unroll
        for (int j = 0; j < 8; j++) { vf[j] = pv[j]; vf[8 + j] = pv[16 + j]; }
        o[t] = wmma_bf16(pa, vf, o[t]);
      }
    }
  }

  // Epilogue: normalize by l and store ctx bf16 in [B,S,768] layout.
#pragma unroll
  for (int t = 0; t < 4; t++) {
    int d = t * 16 + hrow;
#pragma unroll
    for (int rr = 0; rr < 8; rr++) {
      int sq = q0 + wave * 16 + hi * 8 + rr;
      float val = o[t][rr] / lrow[rr];
      Ctx[((size_t)(b * NS + sq)) * ND + h * NDH + d] = (bf16_t)val;
    }
  }
}

// ---------------------------------------------------------------------------
// Kernel 3: output projection + bias + residual (fp32 out).
// Y[m,n] = ctx[m,:] . Wd[n,:] + bd[n] + hidden[m,n]
// ---------------------------------------------------------------------------
__global__ __launch_bounds__(128)
void out_gemm_kernel(const bf16_t* __restrict__ A,      // ctx bf16 [8192,768]
                     const bf16_t* __restrict__ W,      // Wd bf16 [768,768]
                     const float*  __restrict__ bias,
                     const float*  __restrict__ resid,  // hidden fp32
                     float* __restrict__ Y) {
  __shared__ bf16_t As[64][40];
  __shared__ bf16_t Bs[64][40];

  const int m0   = blockIdx.x * 64;
  const int n0   = blockIdx.y * 64;
  const int tid  = threadIdx.x;
  const int wave = tid >> 5;
  const int lane = tid & 31;
  const int hrow = lane & 15;
  const int hi   = lane >> 4;

  v8f acc[4] = {};

  const int r = tid >> 1;
  const int c = (tid & 1) * 16;
  const uint32_t la = (uint32_t)(uintptr_t)&As[r][c];
  const uint32_t lb = (uint32_t)(uintptr_t)&Bs[r][c];

  for (int k0 = 0; k0 < ND; k0 += 32) {
    __syncthreads();
    async_cp32(la, (uint64_t)(uintptr_t)(A + (size_t)(m0 + r) * ND + k0 + c));
    async_cp32(lb, (uint64_t)(uintptr_t)(W + (size_t)(n0 + r) * ND + k0 + c));
    async_wait0();
    __syncthreads();

    v16bf a;
    {
      const bf16_t* pa  = &As[wave * 16 + hrow][hi * 8];
      const bf16_t* pa2 = &As[wave * 16 + hrow][hi * 8 + 16];
#pragma unroll
      for (int j = 0; j < 8; j++) { a[j] = pa[j]; a[8 + j] = pa2[j]; }
    }
#pragma unroll
    for (int t = 0; t < 4; t++) {
      v16bf b;
      const bf16_t* pb = &Bs[t * 16 + hrow][hi * 8];
#pragma unroll
      for (int j = 0; j < 8; j++) { b[j] = pb[j]; b[8 + j] = pb[16 + j]; }
      acc[t] = wmma_bf16(a, b, acc[t]);
    }
  }

#pragma unroll
  for (int t = 0; t < 4; t++) {
    int n = n0 + t * 16 + hrow;
    float bv = bias[n];
#pragma unroll
    for (int rr = 0; rr < 8; rr++) {
      int m = m0 + wave * 16 + hi * 8 + rr;
      size_t idx = (size_t)m * ND + n;
      Y[idx] = acc[t][rr] + bv + resid[idx];
    }
  }
}

// ---------------------------------------------------------------------------
// Kernel 4: row-wise LayerNorm (D=768) -> d_out fp32.
// ---------------------------------------------------------------------------
__global__ __launch_bounds__(256)
void ln_kernel(const float* __restrict__ Y,
               const float* __restrict__ g,
               const float* __restrict__ beta,
               float* __restrict__ out) {
  __shared__ float red[256];
  const int row = blockIdx.x;
  const float* y = Y + (size_t)row * ND;

  float s = 0.0f, s2 = 0.0f;
  for (int i = threadIdx.x; i < ND; i += 256) {
    float v = y[i];
    s += v; s2 += v * v;
  }
  red[threadIdx.x] = s;
  __syncthreads();
  for (int st = 128; st > 0; st >>= 1) {
    if (threadIdx.x < st) red[threadIdx.x] += red[threadIdx.x + st];
    __syncthreads();
  }
  float mean = red[0] * (1.0f / ND);
  __syncthreads();
  red[threadIdx.x] = s2;
  __syncthreads();
  for (int st = 128; st > 0; st >>= 1) {
    if (threadIdx.x < st) red[threadIdx.x] += red[threadIdx.x + st];
    __syncthreads();
  }
  float var = red[0] * (1.0f / ND) - mean * mean;
  float inv = rsqrtf(var + 1e-12f);
  for (int i = threadIdx.x; i < ND; i += 256) {
    out[(size_t)row * ND + i] = (y[i] - mean) * inv * g[i] + beta[i];
  }
}

// ---------------------------------------------------------------------------
// Host launcher
// ---------------------------------------------------------------------------
extern "C" void kernel_launch(void* const* d_in, const int* in_sizes, int n_in,
                              void* d_out, int out_size, void* d_ws, size_t ws_size,
                              hipStream_t stream) {
  const float* hidden = (const float*)d_in[0];
  const float* amask  = (const float*)d_in[1];
  const float* Wq = (const float*)d_in[2];
  const float* bq = (const float*)d_in[3];
  const float* Wk = (const float*)d_in[4];
  const float* bk = (const float*)d_in[5];
  const float* Wv = (const float*)d_in[6];
  const float* bv = (const float*)d_in[7];
  const float* Wd = (const float*)d_in[8];
  const float* bd = (const float*)d_in[9];
  const float* ln_g = (const float*)d_in[10];
  const float* ln_b = (const float*)d_in[11];
  float* out = (float*)d_out;

  // Workspace carve-out (256B aligned).
  size_t off = 0;
  auto carve = [&](size_t bytes) {
    void* p = (char*)d_ws + off;
    off += (bytes + 255) & ~(size_t)255;
    return p;
  };
  const size_t nX = (size_t)NM * ND;       // 6.29M
  const size_t nW = (size_t)ND * ND;       // 0.59M
  bf16_t* Xbf  = (bf16_t*)carve(nX * 2);
  bf16_t* Wqb  = (bf16_t*)carve(nW * 2);
  bf16_t* Wkb  = (bf16_t*)carve(nW * 2);
  bf16_t* Wvb  = (bf16_t*)carve(nW * 2);
  bf16_t* Wdb  = (bf16_t*)carve(nW * 2);
  bf16_t* Qb   = (bf16_t*)carve(nX * 2);
  bf16_t* Kb   = (bf16_t*)carve(nX * 2);
  bf16_t* Vb   = (bf16_t*)carve(nX * 2);
  bf16_t* Ctx  = (bf16_t*)carve(nX * 2);
  float*  Ybuf = (float*)carve(nX * 4);

  // 0) fp32 -> bf16 conversions
  cvt_kernel<<<dim3((nX + 255) / 256), dim3(256), 0, stream>>>(hidden, Xbf, (int)nX);
  cvt_kernel<<<dim3((nW + 255) / 256), dim3(256), 0, stream>>>(Wq, Wqb, (int)nW);
  cvt_kernel<<<dim3((nW + 255) / 256), dim3(256), 0, stream>>>(Wk, Wkb, (int)nW);
  cvt_kernel<<<dim3((nW + 255) / 256), dim3(256), 0, stream>>>(Wv, Wvb, (int)nW);
  cvt_kernel<<<dim3((nW + 255) / 256), dim3(256), 0, stream>>>(Wd, Wdb, (int)nW);

  // 1) QKV projections (Q carries 1/sqrt(64) scale)
  dim3 ggrid(NM / 64, ND / 64, 1);
  qkv_gemm_kernel<<<ggrid, dim3(128), 0, stream>>>(Xbf, Wqb, bq, Qb, 0.125f);
  qkv_gemm_kernel<<<ggrid, dim3(128), 0, stream>>>(Xbf, Wkb, bk, Kb, 1.0f);
  qkv_gemm_kernel<<<ggrid, dim3(128), 0, stream>>>(Xbf, Wvb, bv, Vb, 1.0f);

  // 2) flash attention
  attn_kernel<<<dim3(NS / 64, NH, NB), dim3(128), 0, stream>>>(Qb, Kb, Vb, amask, Ctx);

  // 3) output projection + residual
  out_gemm_kernel<<<ggrid, dim3(128), 0, stream>>>(Ctx, Wdb, bd, hidden, Ybuf);

  // 4) LayerNorm
  ln_kernel<<<dim3(NM), dim3(256), 0, stream>>>(Ybuf, ln_g, ln_b, out);

  (void)in_sizes; (void)n_in; (void)out_size; (void)ws_size;
}
